// RnnMsgAgent_61186104099655
// MI455X (gfx1250) — matrix operand order repaired and checked by reading, not compiled, not run on hardware
//
#include <hip/hip_runtime.h>

// ---------------- problem sizes ----------------
#define B_    4096
#define A_    8
#define H_    512
#define NK_   128
#define NV_   256
#define NQ_   128
#define NACT_ 30
#define ROWS_ (B_ * A_)            // 32768
#define KVQN_ 512                  // key|value|query fused N
#define X2K_  768                  // NV + H
#define W2N_  32                   // 30 padded to 32
#define EVN_  (ROWS_ * NACT_)      // 983040 floats of evidence output

typedef __bf16 bf16_t;
typedef __attribute__((ext_vector_type(16))) __bf16 v16bf;
typedef __attribute__((ext_vector_type(8)))  __bf16 v8bf;
typedef __attribute__((ext_vector_type(8)))  float  v8f;

// ---------------- helpers ----------------
static __device__ __forceinline__ bf16_t f2bf(float f) {
  unsigned u = __float_as_uint(f);
  unsigned r = (u + 0x7FFFu + ((u >> 16) & 1u)) >> 16;   // RNE
  unsigned short s = (unsigned short)r;
  return __builtin_bit_cast(bf16_t, s);
}
static __device__ __forceinline__ float bf2f(bf16_t b) {
  unsigned short s = __builtin_bit_cast(unsigned short, b);
  return __uint_as_float(((unsigned)s) << 16);
}
// monotone float<->uint map for atomic min/max over any-sign floats
static __device__ __forceinline__ unsigned f2mono(float f) {
  unsigned u = __float_as_uint(f);
  return (u & 0x80000000u) ? ~u : (u | 0x80000000u);
}
static __device__ __forceinline__ float mono2f(unsigned m) {
  unsigned u = (m & 0x80000000u) ? (m & 0x7FFFFFFFu) : ~m;
  return __uint_as_float(u);
}
static __device__ __forceinline__ float wsum(float v) {
#pragma unroll
  for (int m = 16; m >= 1; m >>= 1) v += __shfl_xor(v, m, 32);
  return v;
}
static __device__ __forceinline__ float wmin32(float v) {
#pragma unroll
  for (int m = 16; m >= 1; m >>= 1) v = fminf(v, __shfl_xor(v, m, 32));
  return v;
}
static __device__ __forceinline__ float wmax32(float v) {
#pragma unroll
  for (int m = 16; m >= 1; m >>= 1) v = fmaxf(v, __shfl_xor(v, m, 32));
  return v;
}

// ---------------- prep kernels ----------------
__global__ __launch_bounds__(256) void k_f32_to_bf16(const float* __restrict__ src,
                                                     bf16_t* __restrict__ dst, int n) {
  int i = blockIdx.x * 256 + threadIdx.x;
  if (i < n) dst[i] = f2bf(src[i]);
}
// WT[n*K + k] = W[k*N + n]  (bf16 transposed weights, contiguous-K rows)
__global__ __launch_bounds__(256) void k_transpose_bf16(const float* __restrict__ W,
                                                        bf16_t* __restrict__ WT,
                                                        int K, int N) {
  int i = blockIdx.x * 256 + threadIdx.x;
  if (i < K * N) {
    int n = i / K, k = i % K;
    WT[(size_t)n * K + k] = f2bf(W[(size_t)k * N + n]);
  }
}
__global__ __launch_bounds__(256) void k_bias_kvq(const float* __restrict__ bk,
                                                  const float* __restrict__ bv,
                                                  const float* __restrict__ bq,
                                                  float* __restrict__ dst) {
  int n = blockIdx.x * 256 + threadIdx.x;
  if (n < KVQN_) dst[n] = (n < NK_) ? bk[n] : (n < NK_ + NV_) ? bv[n - NK_] : bq[n - NK_ - NV_];
}

// ---------------- WMMA GEMM (software-pipelined) ----------------
// C[ROWS, NTOT] = act( [A0|A1][ROWS, K] @ WT^T + bias )
// A split at compile-time KSPLIT (multiple of 32). WT is [NTOT][K] bf16.
// Each wave: 16 rows x (NT*16) cols; block: 8 waves = 128 rows.
// 2-stage pipeline: chunk c+1's fragments are loaded before chunk c's WMMAs,
// so each v_wmma waits with the next chunk's 2+2*NT loads still in flight.
// OUTMODE: 0 = bf16, 1 = bf16 + relu, 2 = f32 + relu(clip>=0)
template <int OUTMODE, int NT, int K, int KSPLIT, int LDA0, int LDA1, int NTOT>
__global__ __launch_bounds__(256) void wmma_gemm(
    const bf16_t* __restrict__ A0,
    const bf16_t* __restrict__ A1,
    const bf16_t* __restrict__ WT,
    const float* __restrict__ bias, int nbias,
    void* __restrict__ outp) {
  const int lane = threadIdx.x & 31;
  const int wave = threadIdx.x >> 5;                    // wave32
  const int rowBase = blockIdx.x * 128 + wave * 16;     // 8 waves x 16 rows
  const int n0 = blockIdx.y * (NT * 16);
  const bool hi = (lane & 16) != 0;
  const int ml = lane & 15;
  const int aKoff = hi ? 8 : 0;                         // ISA 16-bit A 16x32 layout
  const int bKoff = hi ? 8 : 0;                         // same k order as A (table order)
  const int row = rowBase + ml;

  const bf16_t* ap0 = A0 + (size_t)row * LDA0 + aKoff;
  const bf16_t* ap1 = A1 + (size_t)row * LDA1 + aKoff;
  const bf16_t* wp[NT];
#pragma unroll
  for (int t = 0; t < NT; ++t) {
    wp[t] = WT + (size_t)(n0 + t * 16 + ml) * K + bKoff;
    __builtin_prefetch(wp[t], 0, 1);                    // global_prefetch_b8
  }

  v8f acc[NT] = {};

  struct Frags { v16bf a; v16bf b[NT]; };
  auto loadfrag = [&](int kk) -> Frags {
    Frags f;
    const bf16_t* ac = (kk < KSPLIT) ? (ap0 + kk) : (ap1 + (kk - KSPLIT));
    v8bf alo = *(const v8bf*)(ac);
    v8bf ahi = *(const v8bf*)(ac + 16);
#pragma unroll
    for (int i = 0; i < 8; ++i) { f.a[i] = alo[i]; f.a[8 + i] = ahi[i]; }
#pragma unroll
    for (int t = 0; t < NT; ++t) {
      v8bf blo = *(const v8bf*)(wp[t] + kk);
      v8bf bhi = *(const v8bf*)(wp[t] + kk + 16);
#pragma unroll
      for (int i = 0; i < 8; ++i) { f.b[t][i] = blo[i]; f.b[t][8 + i] = bhi[i]; }
    }
    return f;
  };
  auto domma = [&](const Frags& f) {
#pragma unroll
    for (int t = 0; t < NT; ++t)
      acc[t] = __builtin_amdgcn_wmma_f32_16x16x32_bf16(
          false, f.a, false, f.b[t], (short)0, acc[t], false, false);
  };

  constexpr int NC = K / 32;
  Frags cur = loadfrag(0);
#pragma unroll
  for (int c = 0; c + 1 < NC; ++c) {
    Frags nxt = loadfrag((c + 1) * 32);   // issue next chunk's loads first
    domma(cur);                           // compute current chunk
    cur = nxt;                            // SSA-renamed under full unroll
  }
  domma(cur);

  const int rOff = hi ? 8 : 0;                          // f32 C/D layout
#pragma unroll
  for (int t = 0; t < NT; ++t) {
    int col = n0 + t * 16 + ml;
    float bb = (col < nbias) ? bias[col] : 0.f;
#pragma unroll
    for (int r = 0; r < 8; ++r) {
      int orow = rowBase + rOff + r;
      float v = acc[t][r] + bb;
      if constexpr (OUTMODE >= 1) v = fmaxf(v, 0.f);
      if constexpr (OUTMODE == 2) ((float*)outp)[(size_t)orow * NTOT + col] = v;
      else ((bf16_t*)outp)[(size_t)orow * NTOT + col] = f2bf(v);
    }
  }
}

// ---------------- attention (softmax over A=8 + weighted value sum) --------
// kvq rows: (b*8+a)*512, cols [0,128)=key [128,384)=value [384,512)=query
__global__ __launch_bounds__(256) void attn_kernel(
    const bf16_t* __restrict__ kvq,
    const float* __restrict__ Wattn,   // [128 + 8*128, 8] row-major
    const float* __restrict__ battn,
    bf16_t* __restrict__ appl) {
  __shared__ float skey[A_ * NK_];
  __shared__ float sval[A_ * NV_];
  __shared__ float squery[A_ * NQ_];
  __shared__ float spart[64];
  __shared__ float slog[64];
  __shared__ float sw[64];
  const int b = blockIdx.x;
  const int t = threadIdx.x;
  const bf16_t* base = kvq + (size_t)b * (A_ * KVQN_);
  for (int idx = t; idx < A_ * KVQN_; idx += 256) {
    int a = idx >> 9, c = idx & 511;
    float v = bf2f(base[idx]);
    if (c < NK_) skey[a * NK_ + c] = v;
    else if (c < NK_ + NV_) sval[a * NV_ + (c - NK_)] = v;
    else squery[a * NQ_ + (c - NK_ - NV_)] = v;
  }
  __syncthreads();
  float qp = 0.f;
  if (t < 64) {  // j = col, m = agent (key part) / query row (q part)
    int j = t & 7, m = t >> 3;
    float kp = 0.f;
    const float* wk = Wattn + (size_t)(NQ_ + m * NK_) * A_ + j;
    for (int c = 0; c < NK_; ++c) kp += skey[m * NK_ + c] * wk[(size_t)c * A_];
    spart[m * 8 + j] = kp;
    const float* wq = Wattn + j;
    for (int c = 0; c < NQ_; ++c) qp += squery[m * NQ_ + c] * wq[(size_t)c * A_];
  }
  __syncthreads();
  if (t < 64) {
    int j = t & 7, i = t >> 3;
    float kpart = 0.f;
#pragma unroll
    for (int m = 0; m < 8; ++m) kpart += spart[m * 8 + j];
    slog[i * 8 + j] = qp + kpart + battn[j];
  }
  __syncthreads();
  if (t < 8) {  // softmax row i = t
    float mx = -1e30f;
#pragma unroll
    for (int j = 0; j < 8; ++j) mx = fmaxf(mx, slog[t * 8 + j]);
    float s = 0.f, e[8];
#pragma unroll
    for (int j = 0; j < 8; ++j) { e[j] = __expf(slog[t * 8 + j] - mx); s += e[j]; }
    float inv = 1.f / s;
#pragma unroll
    for (int j = 0; j < 8; ++j) sw[t * 8 + j] = e[j] * inv;
  }
  __syncthreads();
  {
    int i = t >> 5, vb = (t & 31) * 8;
    float accv[8] = {};
#pragma unroll
    for (int j = 0; j < 8; ++j) {
      float w = sw[i * 8 + j];
#pragma unroll
      for (int u = 0; u < 8; ++u) accv[u] += w * sval[j * NV_ + vb + u];
    }
    v8bf ov;
#pragma unroll
    for (int u = 0; u < 8; ++u) ov[u] = f2bf(accv[u]);
    *(v8bf*)(appl + (size_t)(b * A_ + i) * NV_ + vb) = ov;
  }
}

// ---------------- Dempster-Shafer combine, pass 1 -------------------------
// one wave per batch b; lane = action (lanes 30,31 inert)
__global__ __launch_bounds__(256) void combine1(
    const float* __restrict__ ev,      // [ROWS][32]
    float* __restrict__ allev,         // [B][32]
    float* __restrict__ out,           // writes ori_u section
    unsigned* __restrict__ minmax) {
  const int lane = threadIdx.x & 31;
  const int wv = threadIdx.x >> 5;
  const int b = blockIdx.x * 8 + wv;
  const float* e = ev + (size_t)b * (A_ * W2N_);
  const bool act = lane < NACT_;
  float b0 = 0.f, u0 = 1.f;
  float lmin = 1e30f, lmax = -1e30f;
#pragma unroll
  for (int a = 0; a < A_; ++a) {
    float eva = act ? e[a * W2N_ + lane] : 0.f;
    if (act) { lmin = fminf(lmin, eva); lmax = fmaxf(lmax, eva); }
    float S = wsum(act ? (eva + 1.f) : 0.f);          // sum(be+1) over NACT
    float b1 = act ? eva / S : 0.f;
    float u1 = (float)NACT_ / S;
    float sb0 = wsum(b0);
    float sb1 = wsum(b1);
    float sbb = wsum(b0 * b1);
    float denom = 1.f - (sb0 * sb1 - sbb);
    float inv = 1.f / denom;
    float nb = (b0 * b1 + b0 * u1 + b1 * u0) * inv;
    u0 = u0 * u1 * inv;
    b0 = nb;
    if (lane == a) out[EVN_ + a * B_ + b] = (float)NACT_ / S;  // ori_u[a][b]
  }
  float ae = b0 * ((float)NACT_ / u0);                // all_ev
  if (act) allev[b * W2N_ + lane] = ae;
  float aemin = act ? ae : 1e30f, aemax = act ? ae : -1e30f;
  lmin = wmin32(lmin); lmax = wmax32(lmax);
  aemin = wmin32(aemin); aemax = wmax32(aemax);
  if (lane == 0) {
    atomicMin(minmax + 0, f2mono(lmin));
    atomicMin(minmax + 1, f2mono(aemin));
    atomicMax(minmax + 2, f2mono(lmax));
    atomicMax(minmax + 3, f2mono(aemax));
  }
}

// ---------------- combine pass 2: rescale + outputs -----------------------
__global__ __launch_bounds__(256) void combine2(
    const float* __restrict__ ev, const float* __restrict__ allev,
    const unsigned* __restrict__ minmax, float* __restrict__ out) {
  const int lane = threadIdx.x & 31;
  const int wv = threadIdx.x >> 5;
  const int idx = blockIdx.x * 8 + wv;   // row = b*8+a
  const int b = idx >> 3, a = idx & 7;
  const float min_be = mono2f(minmax[0]);
  const float min_re = mono2f(minmax[1]);
  const float max_be = mono2f(minmax[2]);
  const float max_re = mono2f(minmax[3]);
  const float scale = (max_be - min_be) * 0.1f;        // COMM_COEF
  const float inv = 1.f / (max_re - min_re + 0.01f);
  const bool act = lane < NACT_;
  float eva = act ? ev[(size_t)idx * W2N_ + lane] : 0.f;
  float ae  = act ? allev[b * W2N_ + lane] : 0.f;
  float re = scale * (ae - min_re) * inv;
  float comb = eva + re;                               // received_evidence
  if (act) out[(size_t)idx * NACT_ + lane] = eva + comb;  // evidence + received
  float s2 = wsum(act ? (comb + 1.f) : 0.f);
  if (lane == 0) out[EVN_ + ROWS_ + a * B_ + b] = (float)NACT_ / s2;  // com_u[a][b]
}

// ---------------- host ----------------
extern "C" void kernel_launch(void* const* d_in, const int* in_sizes, int n_in,
                              void* d_out, int out_size, void* d_ws, size_t ws_size,
                              hipStream_t stream) {
  const float* hidden = (const float*)d_in[0];
  // d_in[1] = send_target (unused by reference)
  const float* Wk = (const float*)d_in[2];
  const float* bk = (const float*)d_in[3];
  const float* Wv = (const float*)d_in[4];
  const float* bv = (const float*)d_in[5];
  const float* Wq = (const float*)d_in[6];
  const float* bq = (const float*)d_in[7];
  const float* Wattn = (const float*)d_in[8];
  const float* battn = (const float*)d_in[9];
  const float* Wc = (const float*)d_in[10];
  const float* bc = (const float*)d_in[11];
  const float* W2 = (const float*)d_in[12];
  const float* b2 = (const float*)d_in[13];
  float* out = (float*)d_out;

  char* ws = (char*)d_ws;
  size_t off = 0;
  auto alloc = [&](size_t bytes) -> void* {
    void* p = ws + off;
    off += (bytes + 255) & ~(size_t)255;
    return p;
  };
  bf16_t* hid_bf   = (bf16_t*)alloc((size_t)ROWS_ * H_ * 2);       // 32 MB
  bf16_t* kvq_bf   = (bf16_t*)alloc((size_t)ROWS_ * KVQN_ * 2);    // 32 MB
  bf16_t* appl_bf  = (bf16_t*)alloc((size_t)ROWS_ * NV_ * 2);      // 16 MB
  bf16_t* x3_bf    = (bf16_t*)alloc((size_t)ROWS_ * H_ * 2);       // 32 MB
  float*  ev_f     = (float*)alloc((size_t)ROWS_ * W2N_ * 4);      // 4 MB
  float*  allev_f  = (float*)alloc((size_t)B_ * W2N_ * 4);
  bf16_t* WkvqT    = (bf16_t*)alloc((size_t)KVQN_ * H_ * 2);
  bf16_t* WcT      = (bf16_t*)alloc((size_t)H_ * X2K_ * 2);
  bf16_t* W2T      = (bf16_t*)alloc((size_t)W2N_ * H_ * 2);
  float*  bias_kvq = (float*)alloc(KVQN_ * 4);
  unsigned* minmax = (unsigned*)alloc(4 * sizeof(unsigned));
  if (off > ws_size) return;  // workspace too small; bail deterministically

  // init minmax: slots 0,1 = monotone-min start (all ones); 2,3 = zero
  hipMemsetAsync(minmax, 0xFF, 8, stream);
  hipMemsetAsync((char*)minmax + 8, 0x00, 8, stream);
  hipMemsetAsync(W2T, 0x00, (size_t)W2N_ * H_ * 2, stream);  // pad cols 30,31

  // prep: bf16 activations + transposed bf16 weights
  {
    int n = ROWS_ * H_;
    k_f32_to_bf16<<<(n + 255) / 256, 256, 0, stream>>>(hidden, hid_bf, n);
  }
  k_transpose_bf16<<<(H_ * NK_ + 255) / 256, 256, 0, stream>>>(Wk, WkvqT, H_, NK_);
  k_transpose_bf16<<<(H_ * NV_ + 255) / 256, 256, 0, stream>>>(Wv, WkvqT + (size_t)NK_ * H_, H_, NV_);
  k_transpose_bf16<<<(H_ * NQ_ + 255) / 256, 256, 0, stream>>>(Wq, WkvqT + (size_t)(NK_ + NV_) * H_, H_, NQ_);
  k_transpose_bf16<<<(X2K_ * H_ + 255) / 256, 256, 0, stream>>>(Wc, WcT, X2K_, H_);
  k_transpose_bf16<<<(H_ * NACT_ + 255) / 256, 256, 0, stream>>>(W2, W2T, H_, NACT_);
  k_bias_kvq<<<2, 256, 0, stream>>>(bk, bv, bq, bias_kvq);

  // GEMM 1: kvq = hidden @ [Wk|Wv|Wq] + bias   (bf16 out)
  // <OUTMODE, NT, K, KSPLIT, LDA0, LDA1, NTOT>
  wmma_gemm<0, 4, H_, H_, H_, H_, KVQN_>
      <<<dim3(ROWS_ / 128, KVQN_ / 64), 256, 0, stream>>>(
          hid_bf, hid_bf, WkvqT, bias_kvq, KVQN_, kvq_bf);

  // attention: softmax(QK-logits) @ V -> attn_applied (bf16)
  attn_kernel<<<B_, 256, 0, stream>>>(kvq_bf, Wattn, battn, appl_bf);

  // GEMM 2: relu([appl | hidden] @ Wc + bc)  (split-A at k=256, bf16 out)
  wmma_gemm<1, 4, X2K_, NV_, NV_, H_, H_>
      <<<dim3(ROWS_ / 128, H_ / 64), 256, 0, stream>>>(
          appl_bf, hid_bf, WcT, bc, H_, x3_bf);

  // GEMM 3: evidence = clip(x3 @ W2 + b2, 0)  (f32 out, N padded 30->32, NT=2)
  wmma_gemm<2, 2, H_, H_, H_, H_, W2N_>
      <<<dim3(ROWS_ / 128, 1), 256, 0, stream>>>(
          x3_bf, x3_bf, W2T, b2, NACT_, ev_f);

  // Dempster-Shafer combine + global min/max + outputs
  combine1<<<B_ / 8, 256, 0, stream>>>(ev_f, allev_f, out, minmax);
  combine2<<<ROWS_ / 8, 256, 0, stream>>>(ev_f, allev_f, minmax, out);
}
